// Attention_3332894621985
// MI455X (gfx1250) — matrix operand order repaired and checked by reading, not compiled, not run on hardware
//
#include <hip/hip_runtime.h>
#include <hip/hip_bf16.h>

// ---------------------------------------------------------------------------
// ConvS2S attention for MI455X (gfx1250), bf16 WMMA pipeline.
//   B=8, C=1024, T=2048, S=2048, E=1024
// Stage 1: Q = (conved^T @ W_h2e^T + b_h2e + x) * scale            -> bf16 ws
// Stage 2: energy = Q @ K^T                                        -> f32 d_out[0]
// Stage 3: softmax rows (in place on d_out[0])
// Stage 4: attended = A @ V                                        -> bf16 ws
// Stage 5: out2 = conved + (attended @ W_e2h^T + b_e2h)^T          -> f32 d_out[1]
// ---------------------------------------------------------------------------

typedef __attribute__((ext_vector_type(16))) __bf16 v16bf;
typedef __attribute__((ext_vector_type(8)))  __bf16 v8bf;
typedef __attribute__((ext_vector_type(4)))  __bf16 bf16x4;
typedef __attribute__((ext_vector_type(8)))  float  v8f;
typedef __attribute__((ext_vector_type(4)))  float  f32x4;

#define BM   128
#define BN   128
#define BK   64
#define KPAD 8
#define LDK  (BK + KPAD)   // 72 elems = 144 B per LDS row (multiple of 16 B)

// ---- global -> LDS tile loader (converts f32 -> bf16, optional transpose) ---
// Tile is ROWS(=128) x BK. Non-trans: element (r,k) = src[(row0+r)*ld + k0+k].
// Trans:     element (r,k) = src[(k0+k)*ld + row0+r]  (coalesced along r).
template<bool F32, bool TRANS>
__device__ __forceinline__ void load_tile(__bf16* __restrict__ lds,
                                          const void* __restrict__ src,
                                          long ld, long row0, long k0, int tid) {
  if constexpr (!TRANS) {
    constexpr int NV = (128 * BK) / 4;           // 2048 vec4 ops
    for (int i = tid; i < NV; i += 256) {
      int r  = i >> 4;                           // i / (BK/4)
      int kc = (i & 15) << 2;
      bf16x4 o;
      if constexpr (F32) {
        const float* s = (const float*)src + (row0 + r) * ld + (k0 + kc);
        f32x4 v = *(const f32x4*)s;
        o[0] = (__bf16)v[0]; o[1] = (__bf16)v[1];
        o[2] = (__bf16)v[2]; o[3] = (__bf16)v[3];
      } else {
        const __bf16* s = (const __bf16*)src + (row0 + r) * ld + (k0 + kc);
        o = *(const bf16x4*)s;
      }
      *(bf16x4*)&lds[r * LDK + kc] = o;
    }
  } else {
    constexpr int NV = (128 / 4) * BK;           // 2048
    for (int i = tid; i < NV; i += 256) {
      int k = i >> 5;                            // i / (128/4)
      int r = (i & 31) << 2;
      if constexpr (F32) {
        const float* s = (const float*)src + (k0 + k) * ld + (row0 + r);
        f32x4 v = *(const f32x4*)s;
        lds[(r + 0) * LDK + k] = (__bf16)v[0];
        lds[(r + 1) * LDK + k] = (__bf16)v[1];
        lds[(r + 2) * LDK + k] = (__bf16)v[2];
        lds[(r + 3) * LDK + k] = (__bf16)v[3];
      } else {
        const __bf16* s = (const __bf16*)src + (k0 + k) * ld + (row0 + r);
        bf16x4 v = *(const bf16x4*)s;
        lds[(r + 0) * LDK + k] = v[0];
        lds[(r + 1) * LDK + k] = v[1];
        lds[(r + 2) * LDK + k] = v[2];
        lds[(r + 3) * LDK + k] = v[3];
      }
    }
  }
}

// ---- LDS -> WMMA fragment loads (ISA 7.12.2 16-bit layouts, wave32) --------
// A 16x32: lane half=lane>>4, m=lane&15; elem j -> K = (j&7) + (j>>3)*16 + 8*half
__device__ __forceinline__ v16bf load_a_frag(const __bf16* __restrict__ As,
                                             int mbase, int ks, int lane) {
  int half = lane >> 4, m = mbase + (lane & 15);
  const __bf16* p = &As[m * LDK + ks + 8 * half];
  v8bf lo = *(const v8bf*)p;
  v8bf hi = *(const v8bf*)(p + 16);
  return __builtin_shufflevector(lo, hi, 0,1,2,3,4,5,6,7,8,9,10,11,12,13,14,15);
}
// B 32x16: n=lane&15; elem j -> K = (j&7) + (j>>3)*16 + 8*half  (B element (k,n) = Brow[n][k])
__device__ __forceinline__ v16bf load_b_frag(const __bf16* __restrict__ Bs,
                                             int nbase, int ks, int lane) {
  int half = lane >> 4, n = nbase + (lane & 15);
  const __bf16* p = &Bs[n * LDK + ks + 8 * half];
  v8bf lo = *(const v8bf*)p;
  v8bf hi = *(const v8bf*)(p + 16);
  return __builtin_shufflevector(lo, hi, 0,1,2,3,4,5,6,7,8,9,10,11,12,13,14,15);
}

// ---- unified GEMM: Cout[m,n] = sum_k A[m,k] * Brow[n,k], fused epilogues ---
// MODE 0: Q-gen    : bf16 C[m*ldc+n] = (acc + bias[n] + extra[m*eld+n]) * (*scale)
// MODE 1: energy   : f32  C[m*ldc+n] = acc
// MODE 2: bf16 out : bf16 C[m*ldc+n] = acc
// MODE 3: final    : f32  C[n*ldc+m] = acc + bias[n] + extra[n*eld+m]
template<int MODE, bool AF32, bool ATR, bool BF32, bool BTR>
__global__ __launch_bounds__(256) void gemm_wmma_bf16(
    const void* __restrict__ Ap, const void* __restrict__ Bp,
    void* __restrict__ Cp, int K,
    long lda, long ldb, long ldc,
    long a_bstride, long b_bstride, long c_bstride,
    const float* __restrict__ bias, const float* __restrict__ extra,
    long extra_ld, long extra_bstride,
    const float* __restrict__ scale_ptr) {
  __shared__ __bf16 As[BM * LDK];
  __shared__ __bf16 Bs[BN * LDK];

  const int tid  = threadIdx.x;
  const int lane = tid & 31;
  const int wid  = tid >> 5;     // 8 waves
  const int wm   = wid & 3;      // 4 waves along M (4*32 = 128)
  const int wn   = wid >> 2;     // 2 waves along N (2*64 = 128)
  const long row0 = (long)blockIdx.y * BM;
  const long col0 = (long)blockIdx.x * BN;
  const int  bz   = blockIdx.z;

  const void* Ab = AF32 ? (const void*)((const float*)Ap + (long)bz * a_bstride)
                        : (const void*)((const __bf16*)Ap + (long)bz * a_bstride);
  const void* Bb = BF32 ? (const void*)((const float*)Bp + (long)bz * b_bstride)
                        : (const void*)((const __bf16*)Bp + (long)bz * b_bstride);

  v8f acc[2][4];
#pragma unroll
  for (int mi = 0; mi < 2; ++mi)
#pragma unroll
    for (int ni = 0; ni < 4; ++ni)
#pragma unroll
      for (int r = 0; r < 8; ++r) acc[mi][ni][r] = 0.0f;

  for (long k0 = 0; k0 < K; k0 += BK) {
    load_tile<AF32, ATR>(As, Ab, lda, row0, k0, tid);
    load_tile<BF32, BTR>(Bs, Bb, ldb, col0, k0, tid);
    __syncthreads();
#pragma unroll
    for (int ks = 0; ks < BK; ks += 32) {
      v16bf a0 = load_a_frag(As, wm * 32 + 0,  ks, lane);
      v16bf a1 = load_a_frag(As, wm * 32 + 16, ks, lane);
      v16bf bf[4];
#pragma unroll
      for (int ni = 0; ni < 4; ++ni)
        bf[ni] = load_b_frag(Bs, wn * 64 + ni * 16, ks, lane);
#pragma unroll
      for (int ni = 0; ni < 4; ++ni) {
        acc[0][ni] = __builtin_amdgcn_wmma_f32_16x16x32_bf16(
            false, a0, false, bf[ni], (short)0, acc[0][ni], false, false);
        acc[1][ni] = __builtin_amdgcn_wmma_f32_16x16x32_bf16(
            false, a1, false, bf[ni], (short)0, acc[1][ni], false, false);
      }
    }
    __syncthreads();
  }

  // ---- epilogue: C frag element r -> (m = base + 8*half + r, n = lane&15) ---
  float sc = 1.0f;
  if constexpr (MODE == 0) sc = *scale_ptr;
  const float* extra_b = nullptr;
  if constexpr (MODE == 0 || MODE == 3)
    extra_b = extra + (long)bz * extra_bstride;
  __bf16* Cbf = nullptr; float* Cf = nullptr;
  if constexpr (MODE == 0 || MODE == 2)
    Cbf = (__bf16*)Cp + (long)bz * c_bstride;
  else
    Cf  = (float*)Cp + (long)bz * c_bstride;

  const int half = lane >> 4, lc = lane & 15;
#pragma unroll
  for (int mi = 0; mi < 2; ++mi) {
#pragma unroll
    for (int ni = 0; ni < 4; ++ni) {
      const long n  = col0 + wn * 64 + ni * 16 + lc;
      const long mb = row0 + wm * 32 + mi * 16 + 8 * half;
#pragma unroll
      for (int r = 0; r < 8; ++r) {
        const long m = mb + r;
        float v = acc[mi][ni][r];
        if constexpr (MODE == 0) {
          v = (v + bias[n] + extra_b[m * extra_ld + n]) * sc;
          Cbf[m * ldc + n] = (__bf16)v;
        } else if constexpr (MODE == 1) {
          Cf[m * ldc + n] = v;
        } else if constexpr (MODE == 2) {
          Cbf[m * ldc + n] = (__bf16)v;
        } else {
          Cf[n * ldc + m] = v + bias[n] + extra_b[n * extra_ld + m];
        }
      }
    }
  }
}

// ---- softmax over rows of length 2048, in place, fp32 ----------------------
__global__ __launch_bounds__(256) void softmax_rows_2048(float* __restrict__ att) {
  constexpr int SS = 2048;
  float* p = att + (long)blockIdx.x * SS;
  const int tid = threadIdx.x, lane = tid & 31, wid = tid >> 5;
  __shared__ float red[8];

  float v[8];
  float mx = -3.402823466e38f;
#pragma unroll
  for (int i = 0; i < 8; ++i) { v[i] = p[tid + i * 256]; mx = fmaxf(mx, v[i]); }
#pragma unroll
  for (int off = 16; off; off >>= 1) mx = fmaxf(mx, __shfl_xor(mx, off, 32));
  if (lane == 0) red[wid] = mx;
  __syncthreads();
#pragma unroll
  for (int w = 0; w < 8; ++w) mx = fmaxf(mx, red[w]);

  float sum = 0.0f;
#pragma unroll
  for (int i = 0; i < 8; ++i) { v[i] = __expf(v[i] - mx); sum += v[i]; }
#pragma unroll
  for (int off = 16; off; off >>= 1) sum += __shfl_xor(sum, off, 32);
  __syncthreads();
  if (lane == 0) red[wid] = sum;
  __syncthreads();
  float tot = 0.0f;
#pragma unroll
  for (int w = 0; w < 8; ++w) tot += red[w];
  const float inv = 1.0f / tot;
#pragma unroll
  for (int i = 0; i < 8; ++i) p[tid + i * 256] = v[i] * inv;
}

// ---------------------------------------------------------------------------
extern "C" void kernel_launch(void* const* d_in, const int* in_sizes, int n_in,
                              void* d_out, int out_size, void* d_ws, size_t ws_size,
                              hipStream_t stream) {
  constexpr int B = 8, C = 1024, T = 2048, S = 2048, E = 1024;
  (void)in_sizes; (void)n_in; (void)out_size; (void)ws_size;

  const float* conved     = (const float*)d_in[0];  // [B,C,T]
  const float* enc_conved = (const float*)d_in[1];  // [B,S,E]
  const float* enc_comb   = (const float*)d_in[2];  // [B,S,E]
  const float* x          = (const float*)d_in[3];  // [B,T,E]
  const float* scale      = (const float*)d_in[4];  // [1]
  const float* W_h2e      = (const float*)d_in[5];  // [E,C]
  const float* b_h2e      = (const float*)d_in[6];  // [E]
  const float* W_e2h      = (const float*)d_in[7];  // [C,E]
  const float* b_e2h      = (const float*)d_in[8];  // [C]

  float* att  = (float*)d_out;                      // [B,T,S]
  float* out2 = att + (size_t)B * T * S;            // [B,C,T]

  __bf16* Qbf  = (__bf16*)d_ws;                     // [B,T,E] bf16
  __bf16* ATbf = Qbf + (size_t)B * T * E;           // [B,T,E] bf16

  dim3 blk(256);

  // Stage 1: Q = (conved^T @ W_h2e^T + b_h2e + x) * scale   [M=T, N=E, K=C]
  // A = conved (f32, transposed: (t,c) at conved[c*T + t], ld=T)
  gemm_wmma_bf16<0, true, true, true, false><<<dim3(E / BN, T / BM, B), blk, 0, stream>>>(
      conved, W_h2e, Qbf, C,
      /*lda=*/T, /*ldb=*/C, /*ldc=*/E,
      /*a_bs=*/(long)C * T, /*b_bs=*/0, /*c_bs=*/(long)T * E,
      b_h2e, x, /*eld=*/E, /*ebs=*/(long)T * E, scale);

  // Stage 2: energy = Q @ K^T   [M=T, N=S, K=E]  -> f32 d_out
  gemm_wmma_bf16<1, false, false, true, false><<<dim3(S / BN, T / BM, B), blk, 0, stream>>>(
      Qbf, enc_conved, att, E,
      /*lda=*/E, /*ldb=*/E, /*ldc=*/S,
      /*a_bs=*/(long)T * E, /*b_bs=*/(long)S * E, /*c_bs=*/(long)T * S,
      nullptr, nullptr, 0, 0, nullptr);

  // Stage 3: softmax rows (B*T rows of length S)
  softmax_rows_2048<<<dim3(B * T), blk, 0, stream>>>(att);

  // Stage 4: attended = A @ V   [M=T, N=E, K=S]
  // B = enc_comb (f32, transposed: (e,s) at V[s*E + e], ld=E)
  gemm_wmma_bf16<2, true, false, true, true><<<dim3(E / BN, T / BM, B), blk, 0, stream>>>(
      att, enc_comb, ATbf, S,
      /*lda=*/S, /*ldb=*/E, /*ldc=*/E,
      /*a_bs=*/(long)T * S, /*b_bs=*/(long)S * E, /*c_bs=*/(long)T * E,
      nullptr, nullptr, 0, 0, nullptr);

  // Stage 5: out2[b,c,t] = conved[b,c,t] + b_e2h[c] + sum_e attended[t,e]*W_e2h[c,e]
  gemm_wmma_bf16<3, false, false, true, false><<<dim3(C / BN, T / BM, B), blk, 0, stream>>>(
      ATbf, W_e2h, out2, E,
      /*lda=*/E, /*ldb=*/E, /*ldc=*/T,
      /*a_bs=*/(long)T * E, /*b_bs=*/0, /*c_bs=*/(long)C * T,
      b_e2h, conved, /*eld=*/T, /*ebs=*/(long)C * T, nullptr);
}
